// GraphAttentionLayer_68719476736540
// MI455X (gfx1250) — matrix operand order repaired and checked by reading, not compiled, not run on hardware
//
#include <hip/hip_runtime.h>
#include <hip/hip_bf16.h>

// GAT layer, fused flash-softmax formulation for MI455X (gfx1250, wave32).
//
// inputs:  d_in[0]=h (8192x256 f32), d_in[1]=adj (8192x8192 i32),
//          d_in[2]=W (256x256 f32),  d_in[3]=a (512x1 f32)
// output:  d_out = h_prime (8192x256 f32)
// scratch: WT f16 (128KB) | Wh f32 (8MB) | WhT f16 (4MB) | f1,f2 (64KB)  ~12.2MB

typedef __attribute__((ext_vector_type(16))) _Float16 v16h;
typedef __attribute__((ext_vector_type(8)))  _Float16 v8h;
typedef __attribute__((ext_vector_type(8)))  float    v8f;
typedef __attribute__((ext_vector_type(4)))  float    v4f;
typedef __attribute__((ext_vector_type(4)))  int      v4i;

#define N_NODES 8192
#define FDIM    256
#define ALPHA   0.2f
#define JBLK    256   // keys per inner block (32 iterations, halves barrier count)
#define MBLK    16    // query rows per workgroup
#define TPR     16    // threads per tile row (256/MBLK)
#define CPT     16    // e-columns per thread (JBLK/TPR)

union Frag16 { v16h v; v8h h[2]; };

__device__ __forceinline__ v8h cvt8(const float* p) {
  v4f x0 = *(const v4f*)p;
  v4f x1 = *(const v4f*)(p + 4);
  v8h r;
  r[0] = (_Float16)x0[0]; r[1] = (_Float16)x0[1];
  r[2] = (_Float16)x0[2]; r[3] = (_Float16)x0[3];
  r[4] = (_Float16)x1[0]; r[5] = (_Float16)x1[1];
  r[6] = (_Float16)x1[2]; r[7] = (_Float16)x1[3];
  return r;
}

// ---------------- kernel 1: WT[n][k] = (f16) W[k][n] ----------------
__global__ void gat_prep_wt(const float* __restrict__ W, _Float16* __restrict__ WT) {
  int n = blockIdx.x;     // Fout index
  int k = threadIdx.x;    // Fin index
  WT[n * FDIM + k] = (_Float16)W[k * FDIM + n];
}

// ---------------- kernel 2: Wh = h @ W (WMMA f16->f32) ----------------
// Also emits WhT (f16, [Fout][N]) = K-contiguous B operand for the attention GEMM.
__global__ void __launch_bounds__(256)
gat_gemm1(const float* __restrict__ h, const _Float16* __restrict__ WT,
          float* __restrict__ Wh, _Float16* __restrict__ WhT) {
  const int lane = threadIdx.x & 31;
  const int wave = threadIdx.x >> 5;
  const int row0 = blockIdx.x * MBLK;
  const int m  = lane & 15;
  const int hi = lane >> 4;          // lane half selects A/B K-sub-runs
  v8f acc0 = {}, acc1 = {};

  for (int kk = 0; kk < FDIM / 32; ++kk) {
    // A fragment 16x32 f16: lane half 0 -> K {0..15}, half 1 -> K {16..31}
    Frag16 a;
    const float* arow = h + (size_t)(row0 + m) * FDIM + kk * 32 + hi * 16;
    a.h[0] = cvt8(arow);
    a.h[1] = cvt8(arow + 8);
    // B fragment 32x16 f16: N = lane%16, K contiguous (0..15 / 16..31 by lane half)
    int n0 = wave * 32;
    const _Float16* bp0 = WT + (size_t)(n0 + m) * FDIM + kk * 32 + hi * 16;
    Frag16 b0; b0.h[0] = *(const v8h*)bp0; b0.h[1] = *(const v8h*)(bp0 + 8);
    acc0 = __builtin_amdgcn_wmma_f32_16x16x32_f16(false, a.v, false, b0.v,
                                                  (short)0, acc0, false, false);
    const _Float16* bp1 = WT + (size_t)(n0 + 16 + m) * FDIM + kk * 32 + hi * 16;
    Frag16 b1; b1.h[0] = *(const v8h*)bp1; b1.h[1] = *(const v8h*)(bp1 + 8);
    acc1 = __builtin_amdgcn_wmma_f32_16x16x32_f16(false, a.v, false, b1.v,
                                                  (short)0, acc1, false, false);
  }
  // C layout: VGPR r holds (M=r, N=lane) lanes 0-15 and (M=r+8, N=lane-16) lanes 16-31
  int col = wave * 32 + m;
  #pragma unroll
  for (int r = 0; r < 8; ++r) {
    int row = row0 + r + hi * 8;
    float v0 = acc0[r], v1 = acc1[r];
    Wh[(size_t)row * FDIM + col]        = v0;
    Wh[(size_t)row * FDIM + col + 16]   = v1;
    WhT[(size_t)col * N_NODES + row]        = (_Float16)v0;
    WhT[(size_t)(col + 16) * N_NODES + row] = (_Float16)v1;
  }
}

// ---------------- kernel 3: f1 = Wh@a1, f2 = Wh@a2 (wave per row) ----------------
__global__ void __launch_bounds__(256)
gat_f1f2(const float* __restrict__ Wh, const float* __restrict__ a,
         float* __restrict__ f1, float* __restrict__ f2) {
  const int lane = threadIdx.x & 31;
  const int wave = threadIdx.x >> 5;
  const int row  = blockIdx.x * 8 + wave;
  const float* wr = Wh + (size_t)row * FDIM;
  float s1 = 0.f, s2 = 0.f;
  #pragma unroll
  for (int u = 0; u < FDIM / 32; ++u) {
    int c = u * 32 + lane;
    float w = wr[c];
    s1 += w * a[c];
    s2 += w * a[FDIM + c];
  }
  #pragma unroll
  for (int off = 16; off > 0; off >>= 1) {
    s1 += __shfl_xor(s1, off, 32);
    s2 += __shfl_xor(s2, off, 32);
  }
  if (lane == 0) { f1[row] = s1; f2[row] = s2; }
}

// ---------------- kernel 4: fused mask + online softmax + attention@Wh ----------------
__global__ void __launch_bounds__(256)
gat_attn(const int* __restrict__ adj, const float* __restrict__ f1g,
         const float* __restrict__ f2g, const _Float16* __restrict__ WhT,
         float* __restrict__ out) {
  __shared__ __align__(16) float f2_s[N_NODES];         // 32 KB, staged via async-to-LDS
  __shared__ __align__(16) _Float16 p_s[MBLK * JBLK];   // 8 KB probability tile
  __shared__ float redmax[MBLK * TPR];
  __shared__ float redsum[MBLK * TPR];
  __shared__ __align__(16) float corr_s[MBLK];
  __shared__ __align__(16) float scale_s[MBLK];
  __shared__ float m_s[MBLK], l_s[MBLK], f1_s[MBLK];

  const int tid  = threadIdx.x;
  const int lane = tid & 31;
  const int wave = tid >> 5;
  const int row0 = blockIdx.x * MBLK;
  const int m  = lane & 15;
  const int hi = lane >> 4;

  // --- stage f2 (32 KB) into LDS with CDNA5 async global->LDS copies (ASYNCcnt) ---
  #pragma unroll
  for (int p = 0; p < N_NODES / (256 * 4); ++p) {
    int idx = (p * 256 + tid) * 4;                      // 4 floats = 16 B per thread
    unsigned lds = (unsigned)(uintptr_t)(&f2_s[idx]);   // low 32 bits = LDS byte offset
    const float* gp = f2g + idx;
    asm volatile("global_load_async_to_lds_b128 %0, %1, off"
                 :: "v"(lds), "v"(gp) : "memory");
  }
  if (tid < MBLK) {
    f1_s[tid] = f1g[row0 + tid];
    m_s[tid]  = -1e30f;
    l_s[tid]  = 0.f;
  }
  asm volatile("s_wait_asynccnt 0" ::: "memory");
  __syncthreads();

  v8f acc0 = {}, acc1 = {};
  const int er = tid / TPR;              // tile row this thread computes e for
  const int ec = (tid % TPR) * CPT;      // CPT contiguous tile cols

  // adj register double-buffer: loads for block j are issued during block j-JBLK
  const int* aprow = adj + (size_t)(row0 + er) * N_NODES + ec;
  v4i c0 = *(const v4i*)(aprow);
  v4i c1 = *(const v4i*)(aprow + 4);
  v4i c2 = *(const v4i*)(aprow + 8);
  v4i c3 = *(const v4i*)(aprow + 12);

  for (int j0 = 0; j0 < N_NODES; j0 += JBLK) {
    int am[CPT] = {c0[0], c0[1], c0[2], c0[3], c1[0], c1[1], c1[2], c1[3],
                   c2[0], c2[1], c2[2], c2[3], c3[0], c3[1], c3[2], c3[3]};
    int jn = j0 + JBLK;
    if (jn < N_NODES) {                  // prefetch next adj tile (hidden behind phases 1-3)
      const int* np = aprow + jn;
      c0 = *(const v4i*)(np);
      c1 = *(const v4i*)(np + 4);
      c2 = *(const v4i*)(np + 8);
      c3 = *(const v4i*)(np + 12);
    }

    // --- phase 1: e = mask(leaky_relu(f1+f2)), per-thread CPT values ---
    float f1v = f1_s[er];
    float ev[CPT];
    float pm = -1e30f;
    #pragma unroll
    for (int u = 0; u < CPT; ++u) {
      float s = f1v + f2_s[j0 + ec + u];
      s = (s > 0.f) ? s : ALPHA * s;
      s = (am[u] == 0) ? -1e30f : s;
      ev[u] = s;
      pm = fmaxf(pm, s);
    }
    redmax[er * TPR + (tid % TPR)] = pm;
    __syncthreads();
    // --- running max + correction factor ---
    if (tid < MBLK) {
      float mb = redmax[tid * TPR];
      #pragma unroll
      for (int u = 1; u < TPR; ++u) mb = fmaxf(mb, redmax[tid * TPR + u]);
      float mo = m_s[tid];
      float mn = fmaxf(mo, mb);
      corr_s[tid] = __expf(mo - mn);
      m_s[tid] = mn;
    }
    __syncthreads();
    // --- phase 2: p = exp(e - m) -> f16 tile in LDS, partial row sums ---
    float mrow = m_s[er];
    float ps = 0.f;
    #pragma unroll
    for (int u = 0; u < CPT; ++u) {
      float e = ev[u];
      float p = (e < -1e29f) ? 0.f : __expf(e - mrow);
      ps += p;
      p_s[er * JBLK + ec + u] = (_Float16)p;
    }
    redsum[er * TPR + (tid % TPR)] = ps;
    __syncthreads();
    if (tid < MBLK) {
      float t = 0.f;
      #pragma unroll
      for (int u = 0; u < TPR; ++u) t += redsum[tid * TPR + u];
      l_s[tid] = l_s[tid] * corr_s[tid] + t;
    }
    // --- phase 3: rescale accumulators, P(16xJBLK) @ WhT-slice via WMMA ---
    {
      const float* cp = corr_s + hi * 8;     // factor for M = r + hi*8
      v4f cf0 = *(const v4f*)cp;
      v4f cf1 = *(const v4f*)(cp + 4);
      #pragma unroll
      for (int r = 0; r < 4; ++r) { acc0[r] *= cf0[r]; acc1[r] *= cf0[r]; }
      #pragma unroll
      for (int r = 0; r < 4; ++r) { acc0[4 + r] *= cf1[r]; acc1[4 + r] *= cf1[r]; }

      const int n0 = wave * 32;
      #pragma unroll
      for (int kk = 0; kk < JBLK / 32; ++kk) {
        Frag16 a;
        const _Float16* pp = p_s + m * JBLK + kk * 32 + hi * 16;
        a.h[0] = *(const v8h*)pp;           // K run {0..7} (+16 for hi lanes)
        a.h[1] = *(const v8h*)(pp + 8);     // K run {8..15} (+16 for hi lanes)
        const _Float16* bp0 = WhT + (size_t)(n0 + m) * N_NODES + j0 + kk * 32 + hi * 16;
        Frag16 b0; b0.h[0] = *(const v8h*)bp0; b0.h[1] = *(const v8h*)(bp0 + 8);
        acc0 = __builtin_amdgcn_wmma_f32_16x16x32_f16(false, a.v, false, b0.v,
                                                      (short)0, acc0, false, false);
        const _Float16* bp1 = WhT + (size_t)(n0 + 16 + m) * N_NODES + j0 + kk * 32 + hi * 16;
        Frag16 b1; b1.h[0] = *(const v8h*)bp1; b1.h[1] = *(const v8h*)(bp1 + 8);
        acc1 = __builtin_amdgcn_wmma_f32_16x16x32_f16(false, a.v, false, b1.v,
                                                      (short)0, acc1, false, false);
      }
    }
    __syncthreads();  // p_s / red buffers reused next iteration
  }

  // --- epilogue: divide by softmax denominator, store ---
  if (tid < MBLK) scale_s[tid] = 1.f / l_s[tid];
  __syncthreads();
  {
    const float* sp = scale_s + hi * 8;
    v4f s0 = *(const v4f*)sp;
    v4f s1 = *(const v4f*)(sp + 4);
    int col = wave * 32 + m;
    #pragma unroll
    for (int r = 0; r < 8; ++r) {
      float sc = (r < 4) ? s0[r] : s1[r - 4];
      int row = row0 + r + hi * 8;
      out[(size_t)row * FDIM + col]      = acc0[r] * sc;
      out[(size_t)row * FDIM + col + 16] = acc1[r] * sc;
    }
  }
}

extern "C" void kernel_launch(void* const* d_in, const int* in_sizes, int n_in,
                              void* d_out, int out_size, void* d_ws, size_t ws_size,
                              hipStream_t stream) {
  (void)in_sizes; (void)n_in; (void)out_size; (void)ws_size;
  const float* h   = (const float*)d_in[0];
  const int*   adj = (const int*)d_in[1];
  const float* W   = (const float*)d_in[2];
  const float* a   = (const float*)d_in[3];
  float* out = (float*)d_out;

  char* ws = (char*)d_ws;
  _Float16* WT  = (_Float16*)ws;                                    // 128 KB
  float*    Wh  = (float*)(ws + (128 << 10));                       // 8 MB
  _Float16* WhT = (_Float16*)(ws + (128 << 10) + (8 << 20));        // 4 MB
  float*    f1  = (float*)(ws + (128 << 10) + (12 << 20));          // 32 KB
  float*    f2  = f1 + N_NODES;                                     // 32 KB

  gat_prep_wt<<<FDIM, FDIM, 0, stream>>>(W, WT);
  gat_gemm1<<<N_NODES / MBLK, 256, 0, stream>>>(h, WT, Wh, WhT);
  gat_f1f2<<<N_NODES / 8, 256, 0, stream>>>(Wh, a, f1, f2);
  gat_attn<<<N_NODES / MBLK, 256, 0, stream>>>(adj, f1, f2, WhT, out);
}
